// NoiseAwareLoss_29059748725140
// MI455X (gfx1250) — matrix operand order repaired and checked by reading, not compiled, not run on hardware
//
#include <hip/hip_runtime.h>
#include <hip/hip_bf16.h>
#include <math.h>

// ---------------- problem constants (from reference) ----------------
#define NUM_CLASSES 21
#define C_FEAT      2048
#define HF          41
#define WF          41
#define HWF         (HF * WF)       // 1681
#define BATCH       16
#define IMG_H       321
#define IMG_W       321
#define IMG_HW      (IMG_H * IMG_W) // 103041
#define EPS_COS     1e-6f

typedef __attribute__((ext_vector_type(16))) __bf16 v16bf;
typedef __attribute__((ext_vector_type(8)))  float  v8f;

// ---------------------------------------------------------------
// Kernel A: classifier weight norms wn[k] = ||W[k,:]||  (tiny)
// block.x == class index; also zeroes the 4 f64 accumulators.
// ---------------------------------------------------------------
__global__ __launch_bounds__(256)
void nal_wn_kernel(const float* __restrict__ W, float* __restrict__ wn,
                   double* __restrict__ accum) {
    __shared__ float red[256];
    const int k = blockIdx.x;
    float s = 0.f;
    for (int c = threadIdx.x; c < C_FEAT; c += 256) {
        float v = W[k * C_FEAT + c];
        s = fmaf(v, v, s);
    }
    red[threadIdx.x] = s;
    __syncthreads();
    for (int off = 128; off > 0; off >>= 1) {
        if (threadIdx.x < off) red[threadIdx.x] += red[threadIdx.x + off];
        __syncthreads();
    }
    if (threadIdx.x == 0) wn[k] = sqrtf(red[0]);
    if (blockIdx.x == 0 && threadIdx.x < 4) accum[threadIdx.x] = 0.0;
}

// ---------------------------------------------------------------
// Kernel B: corr[b,k,p] = 1 + <f(b,:,p), w(k,:)> / max(||f||*||w||, eps)
// via bf16 WMMA 16x16x32, feature norms fused into the B-tile loads.
// One wave handles one 16-pixel tile x all 21 classes (two M-tiles).
// All loads are UNPREDICATED (indices clamped): out-of-range A rows /
// B columns only produce D entries that are never stored.
// ---------------------------------------------------------------
__global__ __launch_bounds__(128)
void nal_corr_wmma_kernel(const float* __restrict__ F,   // [B, C_FEAT, 41,41]
                          const float* __restrict__ W,   // [21, C_FEAT]
                          const float* __restrict__ wn,  // [21]
                          float* __restrict__ corr)      // [B, 21, 1681]
{
    const int lane = threadIdx.x & 31;
    const int wave = threadIdx.x >> 5;
    const int tile = blockIdx.x * 4 + wave;          // 106 pixel tiles per batch
    if (tile >= (HWF + 15) / 16) return;             // wave-uniform -> EXEC stays full
    const int b     = blockIdx.y;
    const int n     = lane & 15;                     // pixel-within-tile / matrix column
    const int hi    = lane >> 4;                     // which K-half this lane holds
    const int p     = tile * 16 + n;
    const bool pvld = (p < HWF);
    const int  pl   = pvld ? p : (HWF - 1);          // clamped load column

    const float* __restrict__ Fb = F + (size_t)b * C_FEAT * HWF + pl;

    // A-tile rows (lanes 0-15 and 16-31 hold the same rows, different K halves)
    const int m0  = n;                                       // classes 0..15
    const int m1  = 16 + n;                                  // classes 16..31
    const int m1c = (m1 < NUM_CLASSES) ? m1 : 0;             // clamped load row
    const float* __restrict__ W0 = W + (size_t)m0  * C_FEAT;
    const float* __restrict__ W1 = W + (size_t)m1c * C_FEAT;

    v8f acc0 = {};   // classes 0..15
    v8f acc1 = {};   // classes 16..20 (rest garbage, never stored)
    float sumsq = 0.f;

    for (int c0 = 0; c0 < C_FEAT; c0 += 32) {
        // ---- A tiles: elems 0..7 <- K=c0+hi*8.., elems 8..15 <- K=c0+16+hi*8..
        //      both runs are contiguous -> 4x float4 per tile per lane
        const int kA = c0 + hi * 8;
        const int kB = c0 + 16 + hi * 8;
        const float4 wa0 = *(const float4*)(W0 + kA);
        const float4 wa1 = *(const float4*)(W0 + kA + 4);
        const float4 wb0 = *(const float4*)(W0 + kB);
        const float4 wb1 = *(const float4*)(W0 + kB + 4);
        const float4 xa0 = *(const float4*)(W1 + kA);
        const float4 xa1 = *(const float4*)(W1 + kA + 4);
        const float4 xb0 = *(const float4*)(W1 + kB);
        const float4 xb1 = *(const float4*)(W1 + kB + 4);

        v16bf a0, a1;
        a0[0]  = (__bf16)wa0.x; a0[1]  = (__bf16)wa0.y; a0[2]  = (__bf16)wa0.z; a0[3]  = (__bf16)wa0.w;
        a0[4]  = (__bf16)wa1.x; a0[5]  = (__bf16)wa1.y; a0[6]  = (__bf16)wa1.z; a0[7]  = (__bf16)wa1.w;
        a0[8]  = (__bf16)wb0.x; a0[9]  = (__bf16)wb0.y; a0[10] = (__bf16)wb0.z; a0[11] = (__bf16)wb0.w;
        a0[12] = (__bf16)wb1.x; a0[13] = (__bf16)wb1.y; a0[14] = (__bf16)wb1.z; a0[15] = (__bf16)wb1.w;
        a1[0]  = (__bf16)xa0.x; a1[1]  = (__bf16)xa0.y; a1[2]  = (__bf16)xa0.z; a1[3]  = (__bf16)xa0.w;
        a1[4]  = (__bf16)xa1.x; a1[5]  = (__bf16)xa1.y; a1[6]  = (__bf16)xa1.z; a1[7]  = (__bf16)xa1.w;
        a1[8]  = (__bf16)xb0.x; a1[9]  = (__bf16)xb0.y; a1[10] = (__bf16)xb0.z; a1[11] = (__bf16)xb0.w;
        a1[12] = (__bf16)xb1.x; a1[13] = (__bf16)xb1.y; a1[14] = (__bf16)xb1.z; a1[15] = (__bf16)xb1.w;

        // ---- B tile (feature map, bf16 32x16 layout) + fused sum of squares
        float f[16];
#pragma unroll
        for (int g = 0; g < 8; ++g) {
            const int k0 = hi * 8 + 2 * g + (g >= 4 ? 8 : 0);
            f[2 * g]     = Fb[(size_t)(c0 + k0)     * HWF];
            f[2 * g + 1] = Fb[(size_t)(c0 + k0 + 1) * HWF];
        }
        v16bf bm;
#pragma unroll
        for (int j = 0; j < 16; ++j) {
            sumsq = fmaf(f[j], f[j], sumsq);
            bm[j] = (__bf16)f[j];
        }

        acc0 = __builtin_amdgcn_wmma_f32_16x16x32_bf16(false, a0, false, bm,
                                                       (short)0, acc0, false, false);
        acc1 = __builtin_amdgcn_wmma_f32_16x16x32_bf16(false, a1, false, bm,
                                                       (short)0, acc1, false, false);
    }

    // combine the two K-halves of the per-pixel feature norm
    sumsq += __shfl_xor(sumsq, 16);
    const float fnorm = sqrtf(sumsq);

    // D layout: lane holds N=n, M = r + (hi?8:0)
    if (pvld) {
#pragma unroll
        for (int r = 0; r < 8; ++r) {
            const int m = r + (hi ? 8 : 0);
            const float d0 = fmaxf(fnorm * wn[m], EPS_COS);
            corr[((size_t)b * NUM_CLASSES + m) * HWF + p] = 1.f + acc0[r] / d0;
            const int m2 = m + 16;
            if (m2 < NUM_CLASSES) {
                const float d1 = fmaxf(fnorm * wn[m2], EPS_COS);
                corr[((size_t)b * NUM_CLASSES + m2) * HWF + p] = 1.f + acc1[r] / d1;
            }
        }
    }
}

// ---------------------------------------------------------------
// Kernel C: per-pixel log-softmax + on-the-fly bilinear upsample of
// corr (L2-resident) + confidence, reduced into 4 f64 accumulators:
// [0]=sum(match*logp_c) [1]=cnt [2]=wce numer [3]=wce denom
// ---------------------------------------------------------------
__global__ __launch_bounds__(256)
void nal_loss_kernel(const float* __restrict__ yp,   // [B,21,321,321]
                     const int*   __restrict__ ycrf, // [B,321,321]
                     const int*   __restrict__ yret,
                     const float* __restrict__ corr, // [B,21,1681]
                     double* __restrict__ accum)
{
    const int idx = blockIdx.x * 256 + threadIdx.x;
    double l0 = 0.0, l1 = 0.0, l2 = 0.0, l3 = 0.0;

    if (idx < BATCH * IMG_HW) {
        const int b   = idx / IMG_HW;
        const int rem = idx - b * IMG_HW;
        const int y   = rem / IMG_W;
        const int x   = rem - y * IMG_W;

        const int  yc    = ycrf[idx];
        const bool match = (yc == yret[idx]);

        // ---- log-softmax over 21 classes (strided reads, coalesced in x) ----
        float v[NUM_CLASSES];
        float mx = -3.402823466e38f, sumv = 0.f;
#pragma unroll
        for (int k = 0; k < NUM_CLASSES; ++k) {
            v[k] = yp[((size_t)b * NUM_CLASSES + k) * IMG_HW + rem];
            mx = fmaxf(mx, v[k]);
            sumv += v[k];
        }
        float se = 0.f;
#pragma unroll
        for (int k = 0; k < NUM_CLASSES; ++k) se += __expf(v[k] - mx);
        const float logZ     = mx + __logf(se);
        const float logp_c   = v[yc] - logZ;
        const float sum_logp = sumv - (float)NUM_CLASSES * logZ;

        // ---- bilinear sample of corr at half-pixel centers (41 -> 321) ----
        const float scale = (float)HF / (float)IMG_H;
        float sy = fminf(fmaxf((y + 0.5f) * scale - 0.5f, 0.f), (float)(HF - 1));
        float sx = fminf(fmaxf((x + 0.5f) * scale - 0.5f, 0.f), (float)(WF - 1));
        const int y0 = (int)sy, x0 = (int)sx;
        const int y1 = (y0 + 1 < HF) ? y0 + 1 : HF - 1;
        const int x1 = (x0 + 1 < WF) ? x0 + 1 : WF - 1;
        const float wy = sy - (float)y0, wx = sx - (float)x0;

        float maxc = -3.402823466e38f, cstar = 0.f;
        const float* __restrict__ cb = corr + (size_t)b * NUM_CLASSES * HWF;
        for (int k = 0; k < NUM_CLASSES; ++k) {
            const float* __restrict__ ck = cb + (size_t)k * HWF;
            const float c00 = ck[y0 * WF + x0], c01 = ck[y0 * WF + x1];
            const float c10 = ck[y1 * WF + x0], c11 = ck[y1 * WF + x1];
            const float top = c00 + (c01 - c00) * wx;
            const float bot = c10 + (c11 - c10) * wx;
            const float val = top + (bot - top) * wy;
            maxc = fmaxf(maxc, val);
            if (k == yc) cstar = val;
        }
        const float ratio = cstar / maxc;
        const float conf  = ratio * ratio;          // GAMMA = 2
        const float m     = match ? 1.f : 0.f;

        l0 = match ? (double)logp_c : 0.0;
        l1 = match ? 1.0 : 0.0;
        l2 = (double)(conf * sum_logp) - (double)(m * conf * logp_c);
        l3 = (double)((float)NUM_CLASSES * conf) - (double)(m * conf);
    }

    // ---- block reduction (f64), one atomicAdd per block per accumulator ----
    __shared__ double red[256];
    double vals[4] = {l0, l1, l2, l3};
#pragma unroll
    for (int j = 0; j < 4; ++j) {
        red[threadIdx.x] = vals[j];
        __syncthreads();
        for (int off = 128; off > 0; off >>= 1) {
            if (threadIdx.x < off) red[threadIdx.x] += red[threadIdx.x + off];
            __syncthreads();
        }
        if (threadIdx.x == 0) atomicAdd(&accum[j], red[0]);
        __syncthreads();
    }
}

// ---------------------------------------------------------------
// Kernel D: loss = -ce_sum/cnt  +  (-wce_num/wce_den)
// ---------------------------------------------------------------
__global__ void nal_finalize_kernel(const double* __restrict__ accum,
                                    float* __restrict__ out) {
    const double loss_ce  = -accum[0] / accum[1];
    const double loss_wce = -accum[2] / accum[3];
    out[0] = (float)(loss_ce + loss_wce);
}

// ---------------------------------------------------------------
extern "C" void kernel_launch(void* const* d_in, const int* in_sizes, int n_in,
                              void* d_out, int out_size, void* d_ws, size_t ws_size,
                              hipStream_t stream) {
    const float* y_pred = (const float*)d_in[0];
    const int*   ycrf   = (const int*)  d_in[1];
    const int*   yret   = (const int*)  d_in[2];
    const float* fmap   = (const float*)d_in[3];
    const float* cw     = (const float*)d_in[4];
    float*       out    = (float*)d_out;

    char* ws = (char*)d_ws;
    float*  wn    = (float*)ws;            // 21 floats (padded region: 128 B)
    double* accum = (double*)(ws + 128);   // 4 doubles
    float*  corr  = (float*)(ws + 256);    // B*21*1681 floats (~2.26 MB, L2-resident)

    // 1) weight norms + zero accumulators
    nal_wn_kernel<<<NUM_CLASSES, 256, 0, stream>>>(cw, wn, accum);

    // 2) correlation map via bf16 WMMA (fused feature norms)
    const int ptiles = (HWF + 15) / 16;                 // 106
    nal_corr_wmma_kernel<<<dim3((ptiles + 3) / 4, BATCH), 128, 0, stream>>>(
        fmap, cw, wn, corr);

    // 3) fused log-softmax + upsample + confidence + reductions
    const int total = BATCH * IMG_HW;
    nal_loss_kernel<<<(total + 255) / 256, 256, 0, stream>>>(
        y_pred, ycrf, yret, corr, accum);

    // 4) finalize scalar loss
    nal_finalize_kernel<<<1, 1, 0, stream>>>(accum, out);
}
